// LFADSNET_43894565765714
// MI455X (gfx1250) — matrix operand order repaired and checked by reading, not compiled, not run on hardware
//
#include <hip/hip_runtime.h>
#include <hip/hip_bf16.h>

// ---------------------------------------------------------------------------
// LFADS on gfx1250 (MI455X): bf16 WMMA (v_wmma_f32_16x16x32_bf16) for every
// GEMM, fp32 accumulate. Sequential recurrences run as persistent kernels
// with batch split across workgroups (recurrence is independent per batch row).
// x and all weights are pre-converted to bf16 once per launch so the hot
// recurrent loops issue only b128 fragment loads + WMMA + small elementwise.
// Elementwise path uses native CDNA5 transcendentals (v_tanh/v_exp/v_rcp) and
// packed bf16 conversion where the toolchain exposes them.
//
// Input flattening assumption (setup_inputs insertion order, recursive):
//  0 x(B,T,D) 1 y 2 eps_g0(B,G) 3 eps_u(B,T,U)
//  4..7 egf{wih,whh,bih,bhh} 8..11 egb 12..15 ecf 16..19 ecb
//  20..23 con 24..27 gen
//  28,29 g0mean{w,b} 30,31 g0logvar 32,33 umean 34,35 ulogvar
//  36,37 factors 38,39 rfc1 40,41 rfc2 42,43 rfc3
//  44 g0_prior_mu 45 u_prior_mu 46 g0_prior_logkappa 47 u_prior_logkappa
// Output: pred(B,T,D) fp32 then kl scalar at d_out[B*T*D].
// ---------------------------------------------------------------------------

#define B_ 64
#define T_ 512
#define D_ 128
#define EG_ 256
#define EC_ 256
#define C_ 128
#define G_ 256
#define U_ 64
#define FD_ 64
#define CLIP_ 5.0f
#define LOGVAR_MIN_ (-6.90775527898f)
#define LOG2E_ 1.44269504089f

typedef __attribute__((ext_vector_type(16))) __bf16    v16bf;
typedef __attribute__((ext_vector_type(2)))  __bf16    v2bf;
typedef __attribute__((ext_vector_type(8)))  float     v8f;
typedef __attribute__((ext_vector_type(8)))  unsigned  v8u;

// ---- scalar helpers: prefer native CDNA5 ops, guarded fallbacks ----
#if __has_builtin(__builtin_amdgcn_cvt_pk_bf16_f32)
static __device__ __forceinline__ unsigned short f2bf(float f) {
  v2bf p = __builtin_amdgcn_cvt_pk_bf16_f32(f, f);
  return (unsigned short)(__builtin_bit_cast(unsigned, p) & 0xFFFFu);
}
#else
static __device__ __forceinline__ unsigned short f2bf(float f) {
  unsigned u = __builtin_bit_cast(unsigned, f);
  unsigned r = (u + 0x7FFFu + ((u >> 16) & 1u)) >> 16;     // RNE
  return (unsigned short)r;
}
#endif

static __device__ __forceinline__ float bf2f(unsigned short h) {
  unsigned u = ((unsigned)h) << 16;
  return __builtin_bit_cast(float, u);
}

#if __has_builtin(__builtin_amdgcn_tanhf)
#define TANH_(x) __builtin_amdgcn_tanhf(x)
#elif __has_builtin(__builtin_amdgcn_tanh_f32)
#define TANH_(x) __builtin_amdgcn_tanh_f32(x)
#else
#define TANH_(x) tanhf(x)
#endif

static __device__ __forceinline__ float fast_exp(float x) {
#if __has_builtin(__builtin_amdgcn_exp2f)
  return __builtin_amdgcn_exp2f(x * LOG2E_);
#else
  return __expf(x);
#endif
}
static __device__ __forceinline__ float sigmoidf_(float x) {
#if __has_builtin(__builtin_amdgcn_rcpf)
  return __builtin_amdgcn_rcpf(1.0f + fast_exp(-x));
#else
  return 1.0f / (1.0f + fast_exp(-x));
#endif
}

static __device__ __forceinline__ v8f wmma_bf16(v16bf a, v16bf b, v8f c) {
  return __builtin_amdgcn_wmma_f32_16x16x32_bf16(false, a, false, b, (short)0, c,
                                                 false, false);
}

// A fragment (16x32, bf16, row-major, stride ldA elems). m = lane&15,
// K-half by lane>=16; per lane the fragment is two contiguous 16B runs:
// elements k..k+7 (VGPR0..3) and k+16..k+23 (VGPR4..7), k = kbase + kh*8.
// All call sites keep rp 16-byte aligned.
static __device__ __forceinline__ v16bf load_a_bf(const unsigned short* A, int ldA,
                                                  int row0, int kbase, int lane) {
  int m = lane & 15, kh = (lane >> 4) & 1;
  const unsigned short* rp = A + (size_t)(row0 + m) * ldA + kbase + (kh << 3);
  uint4 lo = *(const uint4*)(rp);
  uint4 hi = *(const uint4*)(rp + 16);
  v8u r;
  r[0] = lo.x; r[1] = lo.y; r[2] = lo.z; r[3] = lo.w;
  r[4] = hi.x; r[5] = hi.y; r[6] = hi.z; r[7] = hi.w;
  return __builtin_bit_cast(v16bf, r);
}

// B fragment (32x16) taken from row-major weight W[n][k] (N x K, bf16):
// B[k][n] = W[n][k]. n = lane&15; per lane two 16B runs: k..k+7 and
// k+16..k+23, k = kbase + kh*8 (same K order as A). rp stays 16B aligned.
static __device__ __forceinline__ v16bf load_b_w(const unsigned short* W, int ldW,
                                                 int n0, int kbase, int lane) {
  int n = lane & 15, kh = (lane >> 4) & 1;
  const unsigned short* rp = W + (size_t)(n0 + n) * ldW + kbase + (kh << 3);
  uint4 lo = *(const uint4*)(rp);
  uint4 hi = *(const uint4*)(rp + 16);
  v8u r;
  r[0] = lo.x; r[1] = lo.y; r[2] = lo.z; r[3] = lo.w;
  r[4] = hi.x; r[5] = hi.y; r[6] = hi.z; r[7] = hi.w;
  return __builtin_bit_cast(v16bf, r);
}

// ---------------------------------------------------------------------------
__global__ void cvt_f32_bf16(const float* __restrict__ s,
                             unsigned short* __restrict__ d, int n) {
  int i = blockIdx.x * blockDim.x + threadIdx.x;
  if (i < n) d[i] = f2bf(s[i]);
}
__global__ void zero_kl(float* p) { if (threadIdx.x == 0) *p = 0.0f; }

// ---------------------------------------------------------------------------
// Encoder: 16 blocks = 4 GRUs x 4 batch tiles. 16 waves; wave w owns hidden
// channel tile cj=w (16 cols of H=256). Per step: 36 WMMAs per wave.
struct EncArgs {
  const unsigned short* xbf;           // (B,T,D) bf16
  const unsigned short* wih[4];        // 768x128 bf16
  const unsigned short* whh[4];        // 768x256 bf16
  const float* bih[4];
  const float* bhh[4];
  unsigned short* econ_f;              // [T][B][256] bf16
  unsigned short* econ_b;              // [T][B][256] bf16 (scan order)
  unsigned short* egen;                // [B][512] bf16
};

__global__ __launch_bounds__(512) void enc_kernel(EncArgs a) {
  __shared__ __align__(16) unsigned short h_lds[16 * 256];
  const int gru = blockIdx.x & 3;          // 0 egf, 1 egb, 2 ecf, 3 ecb
  const int b0  = (blockIdx.x >> 2) * 16;
  const int wave = threadIdx.x >> 5;
  const int lane = threadIdx.x & 31;
  const int n0 = wave * 16;
  const int col = n0 + (lane & 15);
  const bool back = (gru & 1);

  for (int i = threadIdx.x; i < 16 * 256; i += blockDim.x) h_lds[i] = 0;
  __syncthreads();

  const unsigned short* wih = a.wih[gru];
  const unsigned short* whh = a.whh[gru];
  const float bir = a.bih[gru][col]       + a.bhh[gru][col];
  const float biz = a.bih[gru][256 + col] + a.bhh[gru][256 + col];
  const float bin = a.bih[gru][512 + col];
  const float bhn = a.bhh[gru][512 + col];

  for (int t = 0; t < T_; ++t) {
    int tt = back ? (T_ - 1 - t) : t;
    const unsigned short* xrow = a.xbf + (size_t)tt * D_;  // + b*T*D via ldA
    v8f accr = {}, accz = {}, accin = {}, acchn = {};
#pragma unroll
    for (int k = 0; k < D_; k += 32) {                     // input projection
      v16bf xa = load_a_bf(xrow, T_ * D_, b0, k, lane);
      accr  = wmma_bf16(xa, load_b_w(wih, D_, n0,       k, lane), accr);
      accz  = wmma_bf16(xa, load_b_w(wih, D_, n0 + 256, k, lane), accz);
      accin = wmma_bf16(xa, load_b_w(wih, D_, n0 + 512, k, lane), accin);
    }
#pragma unroll
    for (int k = 0; k < EG_; k += 32) {                    // recurrent
      v16bf ha = load_a_bf(h_lds, 256, 0, k, lane);
      accr  = wmma_bf16(ha, load_b_w(whh, EG_, n0,       k, lane), accr);
      accz  = wmma_bf16(ha, load_b_w(whh, EG_, n0 + 256, k, lane), accz);
      acchn = wmma_bf16(ha, load_b_w(whh, EG_, n0 + 512, k, lane), acchn);
    }
    float hnew[8];
#pragma unroll
    for (int v = 0; v < 8; ++v) {
      int m = v + ((lane >> 4) << 3);
      float r  = sigmoidf_(accr[v] + bir);
      float z  = sigmoidf_(accz[v] + biz);
      float nn = TANH_(accin[v] + bin + r * (acchn[v] + bhn));
      float ho = bf2f(h_lds[m * 256 + col]);
      hnew[v] = fminf((1.0f - z) * nn + z * ho, CLIP_);
    }
    __syncthreads();                                       // all reads of h done
#pragma unroll
    for (int v = 0; v < 8; ++v) {
      int m = v + ((lane >> 4) << 3);
      unsigned short hb = f2bf(hnew[v]);
      h_lds[m * 256 + col] = hb;
      size_t gidx = ((size_t)t * B_ + b0 + m) * 256 + col;
      if (gru == 2) a.econ_f[gidx] = hb;
      if (gru == 3) a.econ_b[gidx] = hb;
      if (t == T_ - 1) {
        if (gru == 0) a.egen[(size_t)(b0 + m) * 512 + col] = hb;
        if (gru == 1) a.egen[(size_t)(b0 + m) * 512 + 256 + col] = hb;
      }
    }
    __syncthreads();
  }
}

// ---------------------------------------------------------------------------
// g0 / KL(g0) / f0 kernel: one block, 16 waves.
struct G0Args {
  const unsigned short* egen;          // 64x512 bf16
  const unsigned short* w_m;           // 256x512 bf16
  const unsigned short* w_lv;          // 256x512 bf16
  const float *b_m, *b_lv;
  const float* eps_g0;                 // 64x256
  const unsigned short* w_fac;         // 64x256 bf16
  const float* b_fac;
  unsigned short* g_state;             // 64x256 bf16 (g0)
  unsigned short* f_state;             // 64x64  bf16 (f0)
  const float *pmu, *plk;
  float* kl_out;
};

__global__ __launch_bounds__(512) void g0_kernel(G0Args a) {
  __shared__ __align__(16) unsigned short g_lds[64 * 256];
  int wave = threadIdx.x >> 5, lane = threadIdx.x & 31;
  float pmu = *a.pmu, plv = *a.plk;
  float kl = 0.0f;
  for (int task = wave; task < 64; task += 16) {
    int row0 = (task & 3) * 16, n0 = (task >> 2) * 16;
    v8f am = {}, alv = {};
#pragma unroll
    for (int k = 0; k < 512; k += 32) {
      v16bf ea = load_a_bf(a.egen, 512, row0, k, lane);
      am  = wmma_bf16(ea, load_b_w(a.w_m,  512, n0, k, lane), am);
      alv = wmma_bf16(ea, load_b_w(a.w_lv, 512, n0, k, lane), alv);
    }
    int colc = n0 + (lane & 15);
    float bm = a.b_m[colc], blv = a.b_lv[colc];
#pragma unroll
    for (int v = 0; v < 8; ++v) {
      int m = row0 + v + ((lane >> 4) << 3);
      float mu = am[v] + bm;
      float lv = fmaxf(alv[v] + blv, LOGVAR_MIN_);
      float g0 = a.eps_g0[m * 256 + colc] * fast_exp(0.5f * lv) + mu;
      unsigned short gb = f2bf(g0);
      g_lds[m * 256 + colc] = gb;
      a.g_state[m * 256 + colc] = gb;
      kl += 0.5f * (plv - lv + (fast_exp(lv) + (mu - pmu) * (mu - pmu)) * fast_exp(-plv) - 1.0f);
    }
  }
  atomicAdd(a.kl_out, kl * (1.0f / (float)B_));
  __syncthreads();
  for (int task = wave; task < 16; task += 16) {
    int row0 = (task & 3) * 16, n0 = (task >> 2) * 16;
    v8f acc = {};
#pragma unroll
    for (int k = 0; k < 256; k += 32)
      acc = wmma_bf16(load_a_bf(g_lds, 256, row0, k, lane),
                      load_b_w(a.w_fac, 256, n0, k, lane), acc);
    int colc = n0 + (lane & 15);
    float bf_ = a.b_fac[colc];
#pragma unroll
    for (int v = 0; v < 8; ++v) {
      int m = row0 + v + ((lane >> 4) << 3);
      a.f_state[m * 64 + colc] = f2bf(acc[v] + bf_);
    }
  }
}

// ---------------------------------------------------------------------------
// Generator loop: 4 blocks (16 batch rows each), 16 waves, loops t internally.
struct GenArgs {
  const unsigned short* econ_f;        // [T][B][256]
  const unsigned short* econ_b;        // [T][B][256] (scan order; read reversed)
  const unsigned short* g_state;       // 64x256 (g0)
  const unsigned short* f_state;       // 64x64  (f0)
  const float* eps_u;                  // (B,T,U)
  const unsigned short* con_wih;       // 384x576
  const unsigned short* con_whh;       // 384x128
  const float *con_bih, *con_bhh;
  const unsigned short *um_w, *ulv_w;  // 64x128
  const float *um_b, *ulv_b;
  const unsigned short* gen_wih;       // 768x64
  const unsigned short* gen_whh;       // 768x256
  const float *gen_bih, *gen_bhh;
  const unsigned short* w_fac;         // 64x256
  const float* b_fac;
  const unsigned short *w1, *w2, *w3;  // 128x64, 128x128, 128x128
  const float *b1, *b2, *b3;
  const float *pmu, *plk;
  float* pred;                         // (B,T,D) fp32
  float* kl_out;
};

__global__ __launch_bounds__(512) void gen_kernel(GenArgs a) {
  __shared__ __align__(16) unsigned short eaf[16 * 576];   // [ef|eb|f]
  __shared__ __align__(16) unsigned short c_lds[16 * 128];
  __shared__ __align__(16) unsigned short g_lds[16 * 256];
  __shared__ __align__(16) unsigned short u_lds[16 * 64];
  __shared__ __align__(16) unsigned short h1_lds[16 * 128];
  __shared__ __align__(16) unsigned short h2_lds[16 * 128];
  const int b0 = blockIdx.x * 16;
  const int wave = threadIdx.x >> 5, lane = threadIdx.x & 31;
  const int n0w = wave * 16;
  const int colw = n0w + (lane & 15);
  const float pmu = *a.pmu, plv = *a.plk;
  float kl = 0.0f;

  // hoisted loop-invariant biases (per-lane)
  float cbir = 0, cbiz = 0, cbin = 0, cbhn = 0, bb1 = 0, bb2 = 0, bb3 = 0;
  float ubm = 0, ublv = 0, fb_ = 0;
  if (wave < 8) {
    cbir = a.con_bih[colw] + a.con_bhh[colw];
    cbiz = a.con_bih[128 + colw] + a.con_bhh[128 + colw];
    cbin = a.con_bih[256 + colw];
    cbhn = a.con_bhh[256 + colw];
    bb1 = a.b1[colw]; bb2 = a.b2[colw]; bb3 = a.b3[colw];
  }
  if (wave < 4) { ubm = a.um_b[colw]; ublv = a.ulv_b[colw]; fb_ = a.b_fac[colw]; }
  const float gbir = a.gen_bih[colw] + a.gen_bhh[colw];
  const float gbiz = a.gen_bih[256 + colw] + a.gen_bhh[256 + colw];
  const float gbin = a.gen_bih[512 + colw];
  const float gbhn = a.gen_bhh[512 + colw];

  for (int i = threadIdx.x; i < 16 * 128; i += blockDim.x) c_lds[i] = 0;
  for (int i = threadIdx.x; i < 16 * 256; i += blockDim.x)
    g_lds[i] = a.g_state[(size_t)(b0 + (i >> 8)) * 256 + (i & 255)];
  for (int i = threadIdx.x; i < 16 * 64; i += blockDim.x)
    eaf[(i >> 6) * 576 + 512 + (i & 63)] = a.f_state[(size_t)(b0 + (i >> 6)) * 64 + (i & 63)];
  __syncthreads();

  for (int t = 0; t < T_; ++t) {
    // stage ef / eb (reversed) into eaf
    for (int i = threadIdx.x; i < 16 * 256; i += blockDim.x) {
      int m = i >> 8, cc = i & 255;
      eaf[m * 576 + cc]       = a.econ_f[((size_t)t * B_ + b0 + m) * 256 + cc];
      eaf[m * 576 + 256 + cc] = a.econ_b[((size_t)(T_ - 1 - t) * B_ + b0 + m) * 256 + cc];
    }
    __syncthreads();

    // ---- controller GRU (C=128): 8 col tiles on waves 0..7 ----
    float cn[8];
    if (wave < 8) {
      v8f accr = {}, accz = {}, accin = {}, acchn = {};
#pragma unroll
      for (int k = 0; k < 576; k += 32) {
        v16bf ea = load_a_bf(eaf, 576, 0, k, lane);
        accr  = wmma_bf16(ea, load_b_w(a.con_wih, 576, n0w,       k, lane), accr);
        accz  = wmma_bf16(ea, load_b_w(a.con_wih, 576, n0w + 128, k, lane), accz);
        accin = wmma_bf16(ea, load_b_w(a.con_wih, 576, n0w + 256, k, lane), accin);
      }
#pragma unroll
      for (int k = 0; k < 128; k += 32) {
        v16bf ca = load_a_bf(c_lds, 128, 0, k, lane);
        accr  = wmma_bf16(ca, load_b_w(a.con_whh, 128, n0w,       k, lane), accr);
        accz  = wmma_bf16(ca, load_b_w(a.con_whh, 128, n0w + 128, k, lane), accz);
        acchn = wmma_bf16(ca, load_b_w(a.con_whh, 128, n0w + 256, k, lane), acchn);
      }
#pragma unroll
      for (int v = 0; v < 8; ++v) {
        int m = v + ((lane >> 4) << 3);
        float r  = sigmoidf_(accr[v] + cbir);
        float z  = sigmoidf_(accz[v] + cbiz);
        float nn = TANH_(accin[v] + cbin + r * (acchn[v] + cbhn));
        float co = bf2f(c_lds[m * 128 + colw]);
        cn[v] = fminf(fmaxf((1.0f - z) * nn + z * co, 0.0f), CLIP_);
      }
    }
    __syncthreads();
    if (wave < 8) {
#pragma unroll
      for (int v = 0; v < 8; ++v)
        c_lds[(v + ((lane >> 4) << 3)) * 128 + colw] = f2bf(cn[v]);
    }
    __syncthreads();

    // ---- u sample + KL: waves 0..3 (U=64 -> 4 col tiles) ----
    if (wave < 4) {
      v8f am = {}, alv = {};
#pragma unroll
      for (int k = 0; k < 128; k += 32) {
        v16bf ca = load_a_bf(c_lds, 128, 0, k, lane);
        am  = wmma_bf16(ca, load_b_w(a.um_w,  128, n0w, k, lane), am);
        alv = wmma_bf16(ca, load_b_w(a.ulv_w, 128, n0w, k, lane), alv);
      }
#pragma unroll
      for (int v = 0; v < 8; ++v) {
        int m = v + ((lane >> 4) << 3);
        float mu = am[v] + ubm;
        float lv = alv[v] + ublv;
        float eps = a.eps_u[((size_t)(b0 + m) * T_ + t) * U_ + colw];
        u_lds[m * 64 + colw] = f2bf(eps * fast_exp(0.5f * lv) + mu);
        kl += 0.5f * (plv - lv + (fast_exp(lv) + (mu - pmu) * (mu - pmu)) * fast_exp(-plv) - 1.0f);
      }
    }
    __syncthreads();

    // ---- generator GRU (G=256): 16 col tiles, all waves ----
    float gn[8];
    {
      v8f accr = {}, accz = {}, accin = {}, acchn = {};
#pragma unroll
      for (int k = 0; k < 64; k += 32) {
        v16bf ua = load_a_bf(u_lds, 64, 0, k, lane);
        accr  = wmma_bf16(ua, load_b_w(a.gen_wih, 64, n0w,       k, lane), accr);
        accz  = wmma_bf16(ua, load_b_w(a.gen_wih, 64, n0w + 256, k, lane), accz);
        accin = wmma_bf16(ua, load_b_w(a.gen_wih, 64, n0w + 512, k, lane), accin);
      }
#pragma unroll
      for (int k = 0; k < 256; k += 32) {
        v16bf ga = load_a_bf(g_lds, 256, 0, k, lane);
        accr  = wmma_bf16(ga, load_b_w(a.gen_whh, 256, n0w,       k, lane), accr);
        accz  = wmma_bf16(ga, load_b_w(a.gen_whh, 256, n0w + 256, k, lane), accz);
        acchn = wmma_bf16(ga, load_b_w(a.gen_whh, 256, n0w + 512, k, lane), acchn);
      }
#pragma unroll
      for (int v = 0; v < 8; ++v) {
        int m = v + ((lane >> 4) << 3);
        float r  = sigmoidf_(accr[v] + gbir);
        float z  = sigmoidf_(accz[v] + gbiz);
        float nn = TANH_(accin[v] + gbin + r * (acchn[v] + gbhn));
        float go = bf2f(g_lds[m * 256 + colw]);
        gn[v] = fminf(fmaxf((1.0f - z) * nn + z * go, 0.0f), CLIP_);
      }
    }
    __syncthreads();
#pragma unroll
    for (int v = 0; v < 8; ++v)
      g_lds[(v + ((lane >> 4) << 3)) * 256 + colw] = f2bf(gn[v]);
    __syncthreads();

    // ---- factors f = g @ fac^T + b (FD=64): waves 0..3 ----
    if (wave < 4) {
      v8f acc = {};
#pragma unroll
      for (int k = 0; k < 256; k += 32)
        acc = wmma_bf16(load_a_bf(g_lds, 256, 0, k, lane),
                        load_b_w(a.w_fac, 256, n0w, k, lane), acc);
#pragma unroll
      for (int v = 0; v < 8; ++v)
        eaf[(v + ((lane >> 4) << 3)) * 576 + 512 + colw] = f2bf(acc[v] + fb_);
    }
    __syncthreads();

    // ---- readout MLP: h1 = relu(f W1^T), h2 = relu(h1 W2^T), out = h2 W3^T ----
    if (wave < 8) {
      v8f acc = {};
#pragma unroll
      for (int k = 0; k < 64; k += 32)
        acc = wmma_bf16(load_a_bf(eaf + 512, 576, 0, k, lane),
                        load_b_w(a.w1, 64, n0w, k, lane), acc);
#pragma unroll
      for (int v = 0; v < 8; ++v)
        h1_lds[(v + ((lane >> 4) << 3)) * 128 + colw] = f2bf(fmaxf(acc[v] + bb1, 0.0f));
    }
    __syncthreads();
    if (wave < 8) {
      v8f acc = {};
#pragma unroll
      for (int k = 0; k < 128; k += 32)
        acc = wmma_bf16(load_a_bf(h1_lds, 128, 0, k, lane),
                        load_b_w(a.w2, 128, n0w, k, lane), acc);
#pragma unroll
      for (int v = 0; v < 8; ++v)
        h2_lds[(v + ((lane >> 4) << 3)) * 128 + colw] = f2bf(fmaxf(acc[v] + bb2, 0.0f));
    }
    __syncthreads();
    if (wave < 8) {
      v8f acc = {};
#pragma unroll
      for (int k = 0; k < 128; k += 32)
        acc = wmma_bf16(load_a_bf(h2_lds, 128, 0, k, lane),
                        load_b_w(a.w3, 128, n0w, k, lane), acc);
#pragma unroll
      for (int v = 0; v < 8; ++v) {
        int m = v + ((lane >> 4) << 3);
        a.pred[((size_t)(b0 + m) * T_ + t) * D_ + colw] = acc[v] + bb3;
      }
    }
    __syncthreads();
  }
  atomicAdd(a.kl_out, kl * (1.0f / (float)B_));
}

// ---------------------------------------------------------------------------
extern "C" void kernel_launch(void* const* d_in, const int* in_sizes, int n_in,
                              void* d_out, int out_size, void* d_ws, size_t ws_size,
                              hipStream_t stream) {
  (void)in_sizes; (void)n_in; (void)out_size; (void)ws_size;
  const float* x      = (const float*)d_in[0];
  const float* eps_g0 = (const float*)d_in[2];
  const float* eps_u  = (const float*)d_in[3];

  // workspace carve
  char* ws = (char*)d_ws;
  size_t off = 0;
  auto carve = [&](size_t bytes) -> void* {
    void* p = ws + off;
    off += (bytes + 255) & ~(size_t)255;
    return p;
  };
  unsigned short* xbf    = (unsigned short*)carve((size_t)B_ * T_ * D_ * 2);
  unsigned short* econ_f = (unsigned short*)carve((size_t)T_ * B_ * 256 * 2);
  unsigned short* econ_b = (unsigned short*)carve((size_t)T_ * B_ * 256 * 2);
  unsigned short* egen   = (unsigned short*)carve((size_t)B_ * 512 * 2);
  unsigned short* g0s    = (unsigned short*)carve((size_t)B_ * 256 * 2);
  unsigned short* f0s    = (unsigned short*)carve((size_t)B_ * 64 * 2);

  // bf16 conversion jobs (weights + x), executed once per launch
  struct Job { const float* src; unsigned short* dst; int n; };
  Job jobs[24];
  int nj = 0;
  jobs[nj++] = { x, xbf, B_ * T_ * D_ };
  auto wconv = [&](int idx, int n) -> unsigned short* {
    unsigned short* dst = (unsigned short*)carve((size_t)n * 2);
    jobs[nj++] = { (const float*)d_in[idx], dst, n };
    return dst;
  };
  EncArgs ea;
  ea.xbf = xbf;
  for (int g = 0; g < 4; ++g) {
    int base = 4 + 4 * g;
    ea.wih[g] = wconv(base + 0, 768 * 128);
    ea.whh[g] = wconv(base + 1, 768 * 256);
    ea.bih[g] = (const float*)d_in[base + 2];
    ea.bhh[g] = (const float*)d_in[base + 3];
  }
  unsigned short* con_wih = wconv(20, 384 * 576);
  unsigned short* con_whh = wconv(21, 384 * 128);
  unsigned short* gen_wih = wconv(24, 768 * 64);
  unsigned short* gen_whh = wconv(25, 768 * 256);
  unsigned short* w_g0m   = wconv(28, 256 * 512);
  unsigned short* w_g0lv  = wconv(30, 256 * 512);
  unsigned short* w_um    = wconv(32, 64 * 128);
  unsigned short* w_ulv   = wconv(34, 64 * 128);
  unsigned short* w_fac   = wconv(36, 64 * 256);
  unsigned short* w1      = wconv(38, 128 * 64);
  unsigned short* w2      = wconv(40, 128 * 128);
  unsigned short* w3      = wconv(42, 128 * 128);

  float* pred = (float*)d_out;
  float* klp  = pred + (size_t)B_ * T_ * D_;

  zero_kl<<<1, 32, 0, stream>>>(klp);
  for (int j = 0; j < nj; ++j)
    cvt_f32_bf16<<<(jobs[j].n + 255) / 256, 256, 0, stream>>>(jobs[j].src, jobs[j].dst,
                                                              jobs[j].n);

  ea.econ_f = econ_f; ea.econ_b = econ_b; ea.egen = egen;
  enc_kernel<<<16, 512, 0, stream>>>(ea);

  G0Args ga;
  ga.egen = egen; ga.w_m = w_g0m; ga.w_lv = w_g0lv;
  ga.b_m = (const float*)d_in[29]; ga.b_lv = (const float*)d_in[31];
  ga.eps_g0 = eps_g0; ga.w_fac = w_fac; ga.b_fac = (const float*)d_in[37];
  ga.g_state = g0s; ga.f_state = f0s;
  ga.pmu = (const float*)d_in[44]; ga.plk = (const float*)d_in[46];
  ga.kl_out = klp;
  g0_kernel<<<1, 512, 0, stream>>>(ga);

  GenArgs na;
  na.econ_f = econ_f; na.econ_b = econ_b;
  na.g_state = g0s; na.f_state = f0s; na.eps_u = eps_u;
  na.con_wih = con_wih; na.con_whh = con_whh;
  na.con_bih = (const float*)d_in[22]; na.con_bhh = (const float*)d_in[23];
  na.um_w = w_um; na.ulv_w = w_ulv;
  na.um_b = (const float*)d_in[33]; na.ulv_b = (const float*)d_in[35];
  na.gen_wih = gen_wih; na.gen_whh = gen_whh;
  na.gen_bih = (const float*)d_in[26]; na.gen_bhh = (const float*)d_in[27];
  na.w_fac = w_fac; na.b_fac = (const float*)d_in[37];
  na.w1 = w1; na.w2 = w2; na.w3 = w3;
  na.b1 = (const float*)d_in[39]; na.b2 = (const float*)d_in[41];
  na.b3 = (const float*)d_in[43];
  na.pmu = (const float*)d_in[45]; na.plk = (const float*)d_in[47];
  na.pred = pred; na.kl_out = klp;
  gen_kernel<<<4, 512, 0, stream>>>(na);
}
